// CoordinatorGNNSimple_20066087207287
// MI455X (gfx1250) — matrix operand order repaired and checked by reading, not compiled, not run on hardware
//
#include <hip/hip_runtime.h>

typedef _Float16 h2   __attribute__((ext_vector_type(2)));
typedef _Float16 h8   __attribute__((ext_vector_type(8)));
typedef _Float16 v16h __attribute__((ext_vector_type(16)));
typedef float    v8f  __attribute__((ext_vector_type(8)));
typedef int      i4v  __attribute__((vector_size(16)));   // int4 for async b128

#define AS1 __attribute__((address_space(1)))
#define AS3 __attribute__((address_space(3)))

#if defined(__has_builtin)
#if __has_builtin(__builtin_amdgcn_global_load_async_to_lds_b128)
#define HAVE_ASYNC_LDS 1
#endif
#if __has_builtin(__builtin_amdgcn_s_wait_asynccnt)
#define HAVE_WAIT_ASYNC_BUILTIN 1
#endif
#if __has_builtin(__builtin_elementwise_maximumnum)
#define PKMAX(a, b) __builtin_elementwise_maximumnum((a), (b))
#else
#define PKMAX(a, b) __builtin_elementwise_max((a), (b))
#endif
#else
#define PKMAX(a, b) __builtin_elementwise_max((a), (b))
#endif

union H8U { h8 v; h2 p[4]; };

// ---------------------------------------------------------------------------
// Layer 1: out[n][t] = relu(sum_k x[n][k]*W[k][t] + b[t]), stored as f16.
// K is 24 (agents) or 20 (regions). One block per row, 128 threads.
// ---------------------------------------------------------------------------
__global__ void __launch_bounds__(128) l1_kernel(const float* __restrict__ x, int K,
                                                 const float* __restrict__ W,
                                                 const float* __restrict__ b,
                                                 _Float16* __restrict__ out) {
  __shared__ float xs[32];
  const int n = blockIdx.x;
  const int t = threadIdx.x;
  if (t < K) xs[t] = x[n * K + t];
  __syncthreads();
  float s = b[t];
  for (int k = 0; k < K; ++k) s = fmaf(xs[k], W[k * 128 + t], s);
  out[n * 128 + t] = (_Float16)(s > 0.f ? s : 0.f);
}

// ---------------------------------------------------------------------------
// Transpose four 128x128 f32 matrices into f16: Wt[n][k] = W[k][n].
// blockIdx.z selects the matrix. Gives contiguous per-lane WMMA B loads.
// ---------------------------------------------------------------------------
__global__ void __launch_bounds__(256) transpose4_to_f16(
    const float* __restrict__ s0, const float* __restrict__ s1,
    const float* __restrict__ s2, const float* __restrict__ s3,
    _Float16* __restrict__ d0, _Float16* __restrict__ d1,
    _Float16* __restrict__ d2, _Float16* __restrict__ d3) {
  __shared__ float tile[16][17];
  const float* Ws[4] = {s0, s1, s2, s3};
  _Float16* Wd[4] = {d0, d1, d2, d3};
  const float* W = Ws[blockIdx.z];
  _Float16* Wt = Wd[blockIdx.z];
  const int tx = threadIdx.x, ty = threadIdx.y;
  tile[ty][tx] = W[(blockIdx.y * 16 + ty) * 128 + blockIdx.x * 16 + tx];
  __syncthreads();
  Wt[(blockIdx.x * 16 + ty) * 128 + blockIdx.y * 16 + tx] = (_Float16)tile[tx][ty];
}

// ---------------------------------------------------------------------------
// WMMA GEMM: out(Mx128, f16) = act( H(Mx128, f16) @ W(128x128) + bias ),
// W given transposed as Wt[n][k] (f16). One wave computes a 16x16 tile,
// K=128 in four v_wmma_f32_16x16x32_f16 steps.
//
// A-fragment (16-bit A 16x32): lane L<16 holds row M=L, halves 0..15 =
// K[k0+16*hi .. +15] (hi=lane>>4), same k order as the B-fragment.
// B-fragment: lane holds column N=lane&15, halves j = K[k0 + 16*hi + j].
// C/D: vgpr i -> row m0 + 8*hi + i, col n0 + (lane&15).
// ---------------------------------------------------------------------------
__global__ void __launch_bounds__(128) wmma_gemm_1024x128(
    const _Float16* __restrict__ H, const _Float16* __restrict__ Wt,
    const float* __restrict__ bias, int relu, _Float16* __restrict__ out) {
  const int lane = threadIdx.x & 31;
  const int wave = threadIdx.x >> 5;
  const int m0 = (blockIdx.x * 4 + wave) * 16;
  const int n0 = blockIdx.y * 16;
  const int r  = lane & 15;
  const int hi = lane >> 4;

  v8f acc = {};
#pragma unroll
  for (int kc = 0; kc < 4; ++kc) {
    const int k0 = kc * 32;
    const _Float16* ap = H + (m0 + r) * 128 + k0 + hi * 16;
    h8 alo = *(const h8*)ap;          // K = k0 + 16*hi .. +7
    h8 ahi = *(const h8*)(ap + 8);    // K = k0 + 16*hi + 8 .. +7
    v16h a = __builtin_shufflevector(alo, ahi,
                                     0, 1, 2, 3, 4, 5, 6, 7,
                                     8, 9, 10, 11, 12, 13, 14, 15);
    v16h b = *(const v16h*)(Wt + (n0 + r) * 128 + k0 + hi * 16);
    acc = __builtin_amdgcn_wmma_f32_16x16x32_f16(false, a, false, b,
                                                 (short)0, acc, false, false);
  }
  const int col = n0 + r;
  const float bv = (bias != nullptr) ? bias[col] : 0.f;
#pragma unroll
  for (int i = 0; i < 8; ++i) {
    float v = acc[i] + bv;
    if (relu) v = v > 0.f ? v : 0.f;
    out[(m0 + hi * 8 + i) * 128 + col] = (_Float16)v;
  }
}

// ---------------------------------------------------------------------------
// Pairwise scoring: scores[a][r] = sum_h relu(pa[a][h] + pr[r][h]) * w2[h] + b2
// (bs1 already folded into pa). 64x64 pair tile per block, staged into LDS via
// gfx1250 async-to-LDS b128 loads (ASYNCcnt) when available; padded LDS rows;
// 4x4 register blocking; packed-f16 inner loop (v_pk_add/max/fma) with f32
// re-accumulation every 32 h.
// ---------------------------------------------------------------------------
__global__ void __launch_bounds__(256) pairwise_kernel(
    const _Float16* __restrict__ pa, const _Float16* __restrict__ pr,
    const float* __restrict__ ws2, const float* __restrict__ bs2,
    float* __restrict__ out) {
  constexpr int RS = 136;  // padded row stride in halves (272B = 17 banks*16B)
  __shared__ __align__(16) _Float16 sa[64 * RS];
  __shared__ __align__(16) _Float16 sr[64 * RS];
  __shared__ __align__(16) _Float16 sw[128];
  const int tid = threadIdx.x;
  const int a0 = blockIdx.y * 64;
  const int r0 = blockIdx.x * 64;

  // Cooperative tile load: 64 rows x 256B each, in 16B units.
  for (int i = tid; i < 64 * 8; i += 256) {
    const int row = i >> 3, seg = i & 7;
#ifdef HAVE_ASYNC_LDS
    __builtin_amdgcn_global_load_async_to_lds_b128(
        (AS1 i4v*)(pa + (a0 + row) * 128 + seg * 8),
        (AS3 i4v*)(sa + row * RS + seg * 8), 0, 0);
    __builtin_amdgcn_global_load_async_to_lds_b128(
        (AS1 i4v*)(pr + (r0 + row) * 128 + seg * 8),
        (AS3 i4v*)(sr + row * RS + seg * 8), 0, 0);
#else
    *(uint4*)(sa + row * RS + seg * 8) = *(const uint4*)(pa + (a0 + row) * 128 + seg * 8);
    *(uint4*)(sr + row * RS + seg * 8) = *(const uint4*)(pr + (r0 + row) * 128 + seg * 8);
#endif
  }
  if (tid < 128) sw[tid] = (_Float16)ws2[tid];
#ifdef HAVE_ASYNC_LDS
#ifdef HAVE_WAIT_ASYNC_BUILTIN
  __builtin_amdgcn_s_wait_asynccnt(0);
#else
  asm volatile("s_wait_asynccnt 0" ::: "memory");
#endif
#endif
  __syncthreads();

  const int tx = tid & 15, ty = tid >> 4;
  const _Float16* A0 = sa + (ty * 4) * RS;
  const _Float16* R0 = sr + (tx * 4) * RS;
  const h2 hzero = {(_Float16)0.f, (_Float16)0.f};

  float accf[4][4] = {};
#pragma unroll
  for (int cc = 0; cc < 4; ++cc) {       // 4 super-chunks of 32 h
    h2 acc2[4][4];
#pragma unroll
    for (int i = 0; i < 4; ++i)
#pragma unroll
      for (int j = 0; j < 4; ++j) acc2[i][j] = hzero;

#pragma unroll
    for (int c = 0; c < 2; ++c) {        // 2 chunks of 16 h
      const int hb = cc * 32 + c * 16;
      H8U Av[4][2], Rv[4][2], Wv[2];
      Wv[0].v = *(const h8*)(sw + hb);
      Wv[1].v = *(const h8*)(sw + hb + 8);
#pragma unroll
      for (int i = 0; i < 4; ++i) {
        Av[i][0].v = *(const h8*)(A0 + i * RS + hb);
        Av[i][1].v = *(const h8*)(A0 + i * RS + hb + 8);
        Rv[i][0].v = *(const h8*)(R0 + i * RS + hb);
        Rv[i][1].v = *(const h8*)(R0 + i * RS + hb + 8);
      }
#pragma unroll
      for (int i = 0; i < 4; ++i)
#pragma unroll
        for (int j = 0; j < 4; ++j)
#pragma unroll
          for (int s = 0; s < 2; ++s)
#pragma unroll
            for (int k = 0; k < 4; ++k) {
              h2 t = Av[i][s].p[k] + Rv[j][s].p[k];          // v_pk_add_f16
              t = PKMAX(t, hzero);                           // v_pk_max f16
              acc2[i][j] = t * Wv[s].p[k] + acc2[i][j];      // v_pk_fma_f16
            }
    }
#pragma unroll
    for (int i = 0; i < 4; ++i)
#pragma unroll
      for (int j = 0; j < 4; ++j)
        accf[i][j] += (float)acc2[i][j][0] + (float)acc2[i][j][1];
  }

  const float b2 = bs2[0];
#pragma unroll
  for (int i = 0; i < 4; ++i) {
    float4 o;
    o.x = accf[i][0] + b2;
    o.y = accf[i][1] + b2;
    o.z = accf[i][2] + b2;
    o.w = accf[i][3] + b2;
    *(float4*)(out + (a0 + ty * 4 + i) * 1024 + r0 + tx * 4) = o;
  }
}

// ---------------------------------------------------------------------------
extern "C" void kernel_launch(void* const* d_in, const int* in_sizes, int n_in,
                              void* d_out, int out_size, void* d_ws, size_t ws_size,
                              hipStream_t stream) {
  const float* x_agent  = (const float*)d_in[0];
  const float* x_region = (const float*)d_in[1];
  const float* Wa1 = (const float*)d_in[2];
  const float* ba1 = (const float*)d_in[3];
  const float* Wa2 = (const float*)d_in[4];
  const float* ba2 = (const float*)d_in[5];
  const float* Wr1 = (const float*)d_in[6];
  const float* br1 = (const float*)d_in[7];
  const float* Wr2 = (const float*)d_in[8];
  const float* br2 = (const float*)d_in[9];
  const float* Ws1 = (const float*)d_in[10];
  const float* bs1 = (const float*)d_in[11];
  const float* Ws2 = (const float*)d_in[12];
  const float* bs2 = (const float*)d_in[13];
  float* out = (float*)d_out;

  // Workspace layout (f16 elements): ~1.18 MB total.
  _Float16* wsh   = (_Float16*)d_ws;
  _Float16* h1a   = wsh;             // 1024*128
  _Float16* h1r   = wsh + 131072;    // 1024*128
  _Float16* emba  = wsh + 262144;    // 1024*128
  _Float16* embr  = wsh + 393216;    // 1024*128
  _Float16* wa2t  = wsh + 524288;    // 128*128
  _Float16* wr2t  = wsh + 540672;
  _Float16* ws1at = wsh + 557056;
  _Float16* ws1rt = wsh + 573440;
  _Float16* pa = h1a;                // reuse: h1a dead after emba computed
  _Float16* pr = h1r;

  l1_kernel<<<1024, 128, 0, stream>>>(x_agent, 24, Wa1, ba1, h1a);
  l1_kernel<<<1024, 128, 0, stream>>>(x_region, 20, Wr1, br1, h1r);

  dim3 tb(16, 16), tg(8, 8, 4);
  transpose4_to_f16<<<tg, tb, 0, stream>>>(Wa2, Wr2, Ws1, Ws1 + 128 * 128,
                                           wa2t, wr2t, ws1at, ws1rt);

  dim3 gg(16, 8);  // 64 M-tiles / 4 waves-per-block, 8 N-tiles
  wmma_gemm_1024x128<<<gg, 128, 0, stream>>>(h1a, wa2t, ba2, 1, emba);
  wmma_gemm_1024x128<<<gg, 128, 0, stream>>>(h1r, wr2t, br2, 1, embr);
  wmma_gemm_1024x128<<<gg, 128, 0, stream>>>(emba, ws1at, bs1, 0, pa);      // bs1 folded
  wmma_gemm_1024x128<<<gg, 128, 0, stream>>>(embr, ws1rt, nullptr, 0, pr);  // no bias

  dim3 pg(16, 16);  // 64x64 pair tiles
  pairwise_kernel<<<pg, 256, 0, stream>>>(pa, pr, Ws2, bs2, out);
}
